// TorchModernBertLayer_67173288509861
// MI455X (gfx1250) — compile-verified, matrix operand order checked by reading, not run on hardware
//
#include <hip/hip_runtime.h>

// ---------------------------------------------------------------------------
// ModernBERT encoder layer for MI455X (gfx1250, wave32, WMMA).
// - All matmuls: v_wmma_f32_16x16x32_bf16 (bf16 in, fp32 accum).
// - GEMM tiles staged with global_load_async_to_lds_b128 (ASYNCcnt) and
//   double-buffered LDS: one barrier per 64-deep K tile.
// - Weights converted f32->bf16 once (halves repeated weight traffic).
// - Flash-attention streaming softmax: masks read once, no S x S probs tensor.
// ---------------------------------------------------------------------------

typedef __bf16 bf16_t;
typedef __attribute__((ext_vector_type(16))) __bf16 v16bf;
typedef __attribute__((ext_vector_type(8)))  __bf16 v8bf;
typedef __attribute__((ext_vector_type(8)))  float  v8f;

#define B_   2
#define S_   2048
#define H_   768
#define NH_  12
#define HD_  64
#define I_   1152
#define MTOT 4096   // B_*S_

__device__ __forceinline__ bf16_t f2bf(float f) { return (bf16_t)f; }

__device__ __forceinline__ v16bf cat8(v8bf lo, v8bf hi) {
  return __builtin_shufflevector(lo, hi, 0,1,2,3,4,5,6,7,8,9,10,11,12,13,14,15);
}

__device__ __forceinline__ v8f wmma_bf16(v16bf a, v16bf b, v8f c) {
  return __builtin_amdgcn_wmma_f32_16x16x32_bf16(false, a, false, b, (short)0, c,
                                                 false, false);
}

// CDNA5 async copy: global -> LDS, 16 bytes per lane, tracked by ASYNCcnt.
// VDST carries the LDS byte address (low 32 bits of the generic pointer).
__device__ __forceinline__ void async_b128(void* lds, const void* gptr) {
  unsigned off = (unsigned)(size_t)lds;
  asm volatile("global_load_async_to_lds_b128 %0, %1, off"
               :: "v"(off), "v"(gptr) : "memory");
}
__device__ __forceinline__ void wait_async0() {
  asm volatile("s_wait_asynccnt 0" ::: "memory");
}

// ---------------------------------------------------------------------------
// Weight f32 -> bf16 pre-pass (8 elements / thread).
// ---------------------------------------------------------------------------
__global__ __launch_bounds__(256) void w2bf_kernel(
    const float* __restrict__ src, bf16_t* __restrict__ dst)
{
  const size_t i8 = ((size_t)blockIdx.x * 256 + threadIdx.x) * 8;
  const float4 a = *(const float4*)(src + i8);
  const float4 b = *(const float4*)(src + i8 + 4);
  v8bf o;
  o[0] = f2bf(a.x); o[1] = f2bf(a.y); o[2] = f2bf(a.z); o[3] = f2bf(a.w);
  o[4] = f2bf(b.x); o[5] = f2bf(b.y); o[6] = f2bf(b.z); o[7] = f2bf(b.w);
  *(v8bf*)(dst + i8) = o;
}

// ---------------------------------------------------------------------------
// LayerNorm(768) -> bf16.  One row per block, 256 threads (8 waves).
// ---------------------------------------------------------------------------
__global__ __launch_bounds__(256) void ln_bf16_kernel(
    const float* __restrict__ in, const float* __restrict__ g,
    const float* __restrict__ bb, bf16_t* __restrict__ out)
{
  __shared__ float red[8];
  const int row = blockIdx.x, t = threadIdx.x;
  const float* x = in + (size_t)row * H_;
  float v0 = x[t], v1 = x[t + 256], v2 = x[t + 512];
  float s = v0 + v1 + v2;
  #pragma unroll
  for (int m = 16; m >= 1; m >>= 1) s += __shfl_xor(s, m, 32);
  if ((t & 31) == 0) red[t >> 5] = s;
  __syncthreads();
  float tot = 0.f;
  #pragma unroll
  for (int i = 0; i < 8; i++) tot += red[i];
  __syncthreads();
  const float mu = tot * (1.f / 768.f);
  const float d0 = v0 - mu, d1 = v1 - mu, d2 = v2 - mu;
  float q = d0 * d0 + d1 * d1 + d2 * d2;
  #pragma unroll
  for (int m = 16; m >= 1; m >>= 1) q += __shfl_xor(q, m, 32);
  if ((t & 31) == 0) red[t >> 5] = q;
  __syncthreads();
  float qt = 0.f;
  #pragma unroll
  for (int i = 0; i < 8; i++) qt += red[i];
  const float inv = rsqrtf(qt * (1.f / 768.f) + 1e-5f);
  bf16_t* o = out + (size_t)row * H_;
  o[t]       = f2bf(d0 * inv * g[t]       + bb[t]);
  o[t + 256] = f2bf(d1 * inv * g[t + 256] + bb[t + 256]);
  o[t + 512] = f2bf(d2 * inv * g[t + 512] + bb[t + 512]);
}

// ---------------------------------------------------------------------------
// GEMM: C[m,n] = sum_k A[m,k]*W[n,k] (+epilogue).  A,W bf16; C f32.
// 128x128 block tile, BK=64, double-buffered LDS filled by async copies.
// 8 waves in 2(M)x4(N) grid; wave = 64x32 = 4x2 wmma tiles; 16 wmma / K-tile.
// mode: 0 plain | 1 +res | 2 +bias,GELU | 3 +bias,+res
// ---------------------------------------------------------------------------
__global__ __launch_bounds__(256) void gemm_bf16_kernel(
    const bf16_t* __restrict__ A, const bf16_t* __restrict__ W,
    float* __restrict__ C, const float* __restrict__ bias,
    const float* __restrict__ res, int mode, int M, int N, int K)
{
  __shared__ __align__(16) bf16_t Ab[2][128][72];
  __shared__ __align__(16) bf16_t Bb[2][128][72];
  const int t = threadIdx.x;
  const int m0 = blockIdx.y * 128, n0 = blockIdx.x * 128;
  const int wave = t >> 5, lane = t & 31, hf = lane >> 4, lrw = lane & 15;
  const int wm = wave >> 2, wn = wave & 3;
  const int lr = t >> 1, lk = (t & 1) * 32;     // each thread: 32 bf16 per tile

  v8f acc[4][2];
  #pragma unroll
  for (int i = 0; i < 4; i++)
    #pragma unroll
    for (int j = 0; j < 2; j++)
      acc[i][j] = (v8f){0.f,0.f,0.f,0.f,0.f,0.f,0.f,0.f};

  const bf16_t* ag = A + (size_t)(m0 + lr) * K + lk;
  const bf16_t* wg = W + (size_t)(n0 + lr) * K + lk;

  // preload tile 0
  #pragma unroll
  for (int q = 0; q < 4; q++) {
    async_b128(&Ab[0][lr][lk + q * 8], ag + q * 8);
    async_b128(&Bb[0][lr][lk + q * 8], wg + q * 8);
  }

  const int T = K >> 6;
  for (int tt = 0; tt < T; tt++) {
    const int cur = tt & 1;
    wait_async0();        // this thread's buf[cur] landed in LDS
    __syncthreads();      // everyone's landed; buf[cur^1] readers retired
    if (tt + 1 < T) {     // overlap next tile's copies with this tile's wmma
      const bf16_t* agn = ag + (size_t)(tt + 1) * 64;
      const bf16_t* wgn = wg + (size_t)(tt + 1) * 64;
      #pragma unroll
      for (int q = 0; q < 4; q++) {
        async_b128(&Ab[cur ^ 1][lr][lk + q * 8], agn + q * 8);
        async_b128(&Bb[cur ^ 1][lr][lk + q * 8], wgn + q * 8);
      }
    }
    #pragma unroll
    for (int ks = 0; ks < 2; ks++) {
      v16bf af[4], bfv[2];
      #pragma unroll
      for (int mi = 0; mi < 4; mi++) {
        const bf16_t* p = &Ab[cur][wm * 64 + mi * 16 + lrw][ks * 32 + hf * 8];
        af[mi] = cat8(*(const v8bf*)p, *(const v8bf*)(p + 16));
      }
      #pragma unroll
      for (int ni = 0; ni < 2; ni++) {
        const bf16_t* p = &Bb[cur][wn * 32 + ni * 16 + lrw][ks * 32 + hf * 8];
        bfv[ni] = cat8(*(const v8bf*)p, *(const v8bf*)(p + 16));
      }
      #pragma unroll
      for (int mi = 0; mi < 4; mi++)
        #pragma unroll
        for (int ni = 0; ni < 2; ni++)
          acc[mi][ni] = wmma_bf16(af[mi], bfv[ni], acc[mi][ni]);
    }
  }

  #pragma unroll
  for (int mi = 0; mi < 4; mi++)
    #pragma unroll
    for (int ni = 0; ni < 2; ni++)
      #pragma unroll
      for (int r = 0; r < 8; r++) {
        const int row = m0 + wm * 64 + mi * 16 + hf * 8 + r;
        const int col = n0 + wn * 32 + ni * 16 + lrw;
        float v = acc[mi][ni][r];
        if (mode == 1)       v += res[(size_t)row * N + col];
        else if (mode == 2) { v += bias[col];
                              v = 0.5f * v * (1.f + erff(v * 0.70710678118f)); }
        else if (mode == 3)  v += bias[col] + res[(size_t)row * N + col];
        C[(size_t)row * N + col] = v;
      }
}

// ---------------------------------------------------------------------------
// RoPE + split fp32 QKV[4096,2304] -> bf16 Q/K/V [B,NH,S,HD].
// ---------------------------------------------------------------------------
__global__ __launch_bounds__(256) void rope_split_kernel(
    const float* __restrict__ qkv, const int* __restrict__ pos,
    bf16_t* __restrict__ Q, bf16_t* __restrict__ K, bf16_t* __restrict__ V)
{
  const size_t gid = (size_t)blockIdx.x * 256 + threadIdx.x;
  const int pair = (int)(gid & 31);
  const size_t rest = gid >> 5;
  const int s = (int)(rest & (S_ - 1));
  const size_t bh = rest >> 11;
  const int h = (int)(bh % NH_);
  const int b = (int)(bh / NH_);
  const float* row = qkv + ((size_t)b * S_ + s) * (3 * H_);
  const int d0 = pair * 2;
  const float q0 = row[(0 * NH_ + h) * HD_ + d0], q1 = row[(0 * NH_ + h) * HD_ + d0 + 1];
  const float k0 = row[(1 * NH_ + h) * HD_ + d0], k1 = row[(1 * NH_ + h) * HD_ + d0 + 1];
  const float v0 = row[(2 * NH_ + h) * HD_ + d0], v1 = row[(2 * NH_ + h) * HD_ + d0 + 1];
  const int p = pos[(size_t)b * S_ + s];
  const float freq = __expf(-((float)d0 * (1.f / 64.f)) * 9.210340371976184f); // 10000^(-d0/64)
  float sn, cs;
  __sincosf((float)p * freq, &sn, &cs);
  const size_t o = (bh * S_ + s) * HD_ + d0;
  Q[o]     = f2bf(q0 * cs - q1 * sn);
  Q[o + 1] = f2bf(q1 * cs + q0 * sn);
  K[o]     = f2bf(k0 * cs - k1 * sn);
  K[o + 1] = f2bf(k1 * cs + k0 * sn);
  V[o]     = f2bf(v0);
  V[o + 1] = f2bf(v1);
}

// ---------------------------------------------------------------------------
// Flash attention.  Block = one (b,h) x 128 q-rows, 8 waves x 16 q-rows.
// Keys streamed in chunks of 32: QK^T (4 wmma) -> online softmax (16-lane
// shfl reductions) -> P via per-wave LDS (s_wait_dscnt) -> PV (4 wmma).
// ---------------------------------------------------------------------------
__global__ __launch_bounds__(256) void flash_attn_kernel(
    const bf16_t* __restrict__ Q, const bf16_t* __restrict__ K,
    const bf16_t* __restrict__ V, const float* __restrict__ am,
    const float* __restrict__ sm, bf16_t* __restrict__ ctx)
{
  __shared__ __align__(16) bf16_t Ksh[32][72];      // [key][d]
  __shared__ __align__(16) bf16_t Vsh[64][40];      // [d][key]  (transposed)
  __shared__ __align__(16) bf16_t Psh[8][16][40];   // per-wave P tile

  const int t = threadIdx.x, wave = t >> 5, lane = t & 31;
  const int hf = lane >> 4, lrw = lane & 15;
  const int blk = blockIdx.x;
  const int qt = blk & 15;
  const int bh = blk >> 4;
  const int h = bh % NH_;
  const int b = bh / NH_;
  const int q0 = qt * 128;

  v16bf qa[2];                                       // Q fragments, K-dim = HD
  {
    const bf16_t* qrow = Q + ((size_t)bh * S_ + q0 + wave * 16 + lrw) * HD_;
    #pragma unroll
    for (int ks = 0; ks < 2; ks++) {
      const bf16_t* p = qrow + ks * 32 + hf * 8;
      qa[ks] = cat8(*(const v8bf*)p, *(const v8bf*)(p + 16));
    }
  }

  float mrow[8], lsum[8];
  v8f oacc[4];
  #pragma unroll
  for (int r = 0; r < 8; r++) { mrow[r] = -3.0e38f; lsum[r] = 0.f; }
  #pragma unroll
  for (int nj = 0; nj < 4; nj++) oacc[nj] = (v8f){0,0,0,0,0,0,0,0};

  const float* amb = am + (size_t)b * S_ * S_;
  const float* smb = sm + (size_t)b * S_ * S_;
  const int ldkey = t >> 3, lddd = (t & 7) * 8;

  for (int kc = 0; kc < S_; kc += 32) {
    __syncthreads();
    {
      const bf16_t* kg = K + ((size_t)bh * S_ + kc + ldkey) * HD_ + lddd;
      *(v8bf*)&Ksh[ldkey][lddd] = *(const v8bf*)kg;
      const bf16_t* vg = V + ((size_t)bh * S_ + kc + ldkey) * HD_ + lddd;
      v8bf vv = *(const v8bf*)vg;
      #pragma unroll
      for (int i = 0; i < 8; i++) Vsh[lddd + i][ldkey] = vv[i];   // transpose
    }
    __syncthreads();

    v8f sc[2];
    sc[0] = (v8f){0,0,0,0,0,0,0,0};
    sc[1] = (v8f){0,0,0,0,0,0,0,0};
    #pragma unroll
    for (int j = 0; j < 2; j++)
      #pragma unroll
      for (int ks = 0; ks < 2; ks++) {
        const bf16_t* p = &Ksh[j * 16 + lrw][ks * 32 + hf * 8];
        v16bf kf = cat8(*(const v8bf*)p, *(const v8bf*)(p + 16));
        sc[j] = wmma_bf16(qa[ks], kf, sc[j]);
      }

    float fac[8];
    #pragma unroll
    for (int r = 0; r < 8; r++) {
      const int qr = q0 + wave * 16 + hf * 8 + r;
      const size_t mo = (size_t)qr * S_ + kc + lrw;
      float s0 = sc[0][r] * 0.125f + amb[mo]      + smb[mo];
      float s1 = sc[1][r] * 0.125f + amb[mo + 16] + smb[mo + 16];
      float mx = fmaxf(s0, s1);
      #pragma unroll
      for (int d = 8; d >= 1; d >>= 1) mx = fmaxf(mx, __shfl_xor(mx, d, 32));
      const float mnew = fmaxf(mrow[r], mx);
      const float p0 = __expf(s0 - mnew);
      const float p1 = __expf(s1 - mnew);
      float ps = p0 + p1;
      #pragma unroll
      for (int d = 8; d >= 1; d >>= 1) ps += __shfl_xor(ps, d, 32);
      fac[r] = __expf(mrow[r] - mnew);
      lsum[r] = lsum[r] * fac[r] + ps;
      mrow[r] = mnew;
      Psh[wave][hf * 8 + r][lrw]      = f2bf(p0);
      Psh[wave][hf * 8 + r][16 + lrw] = f2bf(p1);
    }
    #pragma unroll
    for (int nj = 0; nj < 4; nj++)
      #pragma unroll
      for (int r = 0; r < 8; r++) oacc[nj][r] *= fac[r];

    // P produced & consumed within the same wave: LDS-counter wait suffices.
    asm volatile("s_wait_dscnt 0" ::: "memory");

    const bf16_t* pp = &Psh[wave][lrw][hf * 8];
    v16bf pa = cat8(*(const v8bf*)pp, *(const v8bf*)(pp + 16));
    #pragma unroll
    for (int nj = 0; nj < 4; nj++) {
      const bf16_t* vp = &Vsh[nj * 16 + lrw][hf * 8];
      v16bf vf = cat8(*(const v8bf*)vp, *(const v8bf*)(vp + 16));
      oacc[nj] = wmma_bf16(pa, vf, oacc[nj]);
    }
  }

  #pragma unroll
  for (int nj = 0; nj < 4; nj++)
    #pragma unroll
    for (int r = 0; r < 8; r++) {
      const int qr = q0 + wave * 16 + hf * 8 + r;
      const int d = nj * 16 + lrw;
      const float ov = oacc[nj][r] / lsum[r];
      ctx[(((size_t)b * S_ + qr) * NH_ + h) * HD_ + d] = f2bf(ov);  // [B,S,NH,HD]
    }
}

// ---------------------------------------------------------------------------
// GLU gating: gated[m,i] = h[m,2i] * h[m,2i+1]   (h already GELU'd), -> bf16
// ---------------------------------------------------------------------------
__global__ __launch_bounds__(256) void gate_kernel(
    const float* __restrict__ hact, bf16_t* __restrict__ gated)
{
  const size_t gid = (size_t)blockIdx.x * 256 + threadIdx.x;
  const size_t m = gid / I_;
  const int i = (int)(gid % I_);
  const float2 p = *(const float2*)(hact + m * (2 * I_) + 2 * i);
  gated[gid] = f2bf(p.x * p.y);
}

// ---------------------------------------------------------------------------
extern "C" void kernel_launch(void* const* d_in, const int* in_sizes, int n_in,
                              void* d_out, int out_size, void* d_ws, size_t ws_size,
                              hipStream_t stream)
{
  (void)in_sizes; (void)n_in; (void)out_size; (void)ws_size;
  const float* hidden_states = (const float*)d_in[0];
  const float* attn_mask     = (const float*)d_in[1];
  const float* slide_mask    = (const float*)d_in[2];
  const int*   position_ids  = (const int*)d_in[3];
  const float* qkv_w   = (const float*)d_in[4];
  const float* o_w     = (const float*)d_in[5];
  const float* attn_nw = (const float*)d_in[6];
  const float* attn_nb = (const float*)d_in[7];
  const float* mlp_nw  = (const float*)d_in[8];
  const float* mlp_nb  = (const float*)d_in[9];
  const float* wi_w    = (const float*)d_in[10];
  const float* wi_b    = (const float*)d_in[11];
  const float* wo_w    = (const float*)d_in[12];
  const float* wo_b    = (const float*)d_in[13];
  float* out = (float*)d_out;

  // workspace layout
  char* w = (char*)d_ws;
  float*  big    = (float*)(w);                                  // 4096*2304 f32 (QKV, then GELU(h))
  bf16_t* xln    = (bf16_t*)(w + 37748736ull);                   // 4096*768 bf16 (x, then y)
  bf16_t* Qb     = (bf16_t*)(w + 37748736ull + 6291456ull * 1);  // [B,NH,S,HD]
  bf16_t* Kb     = (bf16_t*)(w + 37748736ull + 6291456ull * 2);
  bf16_t* Vb     = (bf16_t*)(w + 37748736ull + 6291456ull * 3);
  bf16_t* Ctx    = (bf16_t*)(w + 37748736ull + 6291456ull * 4);  // [B,S,NH,HD]
  float*  hid    = (float*)(w + 37748736ull + 6291456ull * 5);   // 4096*768 f32
  bf16_t* gated  = (bf16_t*)(w + 81788928ull);                   // 4096*1152 bf16
  bf16_t* qkv_wb = (bf16_t*)(w + 91226112ull);                   // 2304*768
  bf16_t* o_wb   = (bf16_t*)(w + 94765056ull);                   // 768*768
  bf16_t* wi_wb  = (bf16_t*)(w + 95944704ull);                   // 2304*768
  bf16_t* wo_wb  = (bf16_t*)(w + 99483648ull);                   // 768*1152

  // 0) weight pre-conversion to bf16 (once per launch; removes f32->bf16 from
  //    GEMM inner loops and halves repeated weight traffic)
  w2bf_kernel<<<864, 256, 0, stream>>>(qkv_w, qkv_wb);
  w2bf_kernel<<<288, 256, 0, stream>>>(o_w,   o_wb);
  w2bf_kernel<<<864, 256, 0, stream>>>(wi_w,  wi_wb);
  w2bf_kernel<<<432, 256, 0, stream>>>(wo_w,  wo_wb);

  // 1) LN1
  ln_bf16_kernel<<<MTOT, 256, 0, stream>>>(hidden_states, attn_nw, attn_nb, xln);
  // 2) QKV GEMM  [4096,768]x[768,2304]
  gemm_bf16_kernel<<<dim3(18, 32), 256, 0, stream>>>(
      xln, qkv_wb, big, nullptr, nullptr, 0, MTOT, 3 * H_, H_);
  // 3) RoPE + split
  rope_split_kernel<<<6144, 256, 0, stream>>>(big, position_ids, Qb, Kb, Vb);
  // 4) Flash attention (B*NH*16 q-tiles = 384 blocks)
  flash_attn_kernel<<<384, 256, 0, stream>>>(Qb, Kb, Vb, attn_mask, slide_mask, Ctx);
  // 5) O-proj + residual
  gemm_bf16_kernel<<<dim3(6, 32), 256, 0, stream>>>(
      Ctx, o_wb, hid, nullptr, hidden_states, 1, MTOT, H_, H_);
  // 6) LN2
  ln_bf16_kernel<<<MTOT, 256, 0, stream>>>(hid, mlp_nw, mlp_nb, xln);
  // 7) Wi GEMM + bias + exact GELU
  gemm_bf16_kernel<<<dim3(18, 32), 256, 0, stream>>>(
      xln, wi_wb, big, wi_b, nullptr, 2, MTOT, 2 * I_, H_);
  // 8) gating
  gate_kernel<<<18432, 256, 0, stream>>>(big, gated);
  // 9) Wo GEMM + bias + residual -> out
  gemm_bf16_kernel<<<dim3(6, 32), 256, 0, stream>>>(
      gated, wo_wb, out, wo_b, hid, 3, MTOT, H_, I_);
}
